// GDSAFusion_12214886990422
// MI455X (gfx1250) — matrix operand order repaired and checked by reading, not compiled, not run on hardware
//
#include <hip/hip_runtime.h>
#include <math.h>

// ---------------------------------------------------------------------------
// GDSA fusion block for MI455X (gfx1250, wave32).
// All 1x1 convs -> WMMA GEMM (v_wmma_f32_16x16x32_f16, f32 accum):
//   * block of MW waves shares a 32(K)x64(N) activation tile in LDS, staged by
//     CDNA5 async global->LDS DMA (global_load_async_to_lds_b128, ASYNCcnt),
//     DOUBLE-BUFFERED so the DMA of tile k+1 overlaps WMMA on tile k
//   * BN-fold / bias / activation / residual fused in the epilogue
// Depthwise convs, LN, GRN, SE, softmax+RPB, NA-AV as bandwidth-bound kernels.
// ---------------------------------------------------------------------------

typedef __attribute__((ext_vector_type(16))) _Float16 v16h;
typedef __attribute__((ext_vector_type(8)))  float    v8f;

#define HWIMG 3136
#define HDIM  56

// CDNA5 async global->LDS staging (ASYNCcnt-tracked). Set to 0 to fall back
// to plain b128 load + ds_store staging.
#define USE_ASYNC_LDS 1

__device__ __forceinline__ float geluf(float x) {
    return 0.5f * x * (1.0f + erff(x * 0.70710678118654752f));
}
__device__ __forceinline__ float act_apply(float v, int act) {
    if (act == 1) return geluf(v);
    if (act == 2) return v / (1.0f + expf(-v));        // silu
    if (act == 3) return 1.0f / (1.0f + expf(-v));     // sigmoid
    return v;
}

// ---------------------------------------------------------------------------
// WMMA GEMM for NCHW 1x1 conv:
//   Y[z, m, p] = act(sA[m]*(sum_k W[m,k]*X[z,k,p]) + sB[m]) (+ resid)
// grid = (Cout/(16*MW), NP/64, Z), block = 32*MW (each wave: 16x64 tile).
//   A (16x32 f16)  = weight tile: lanes0-15 row=lr K{0..7,16..23},
//                                 lanes16-31 row=lr K{8..15,24..31}
//   B (32x16 f16)  = activation tile: lanes0-15 K{0..7,16..23}, lanes16-31 K{8..15,24..31},
//                    col = lr  (fragments assembled from the shared LDS tile)
//   C/D (16x16 f32)= vgpr v -> M = v + 8*half, lane -> N = lr
// ---------------------------------------------------------------------------
__global__ void k_gemm1x1(const float* __restrict__ X, long xStride,
                          const float* __restrict__ Wm, long wStride,
                          float* __restrict__ Y, long yStride,
                          const float* __restrict__ sA, const float* __restrict__ sB,
                          const float* __restrict__ resid, long rStride,
                          int Cin, int NP, int act) {
    __shared__ float tileB[2][32 * 64];   // double-buffered K-major [32][64] f32

    const int tid  = threadIdx.x;
    const int lane = tid & 31;
    const int wv   = tid >> 5;
    const int nthr = blockDim.x;
    const int MW   = nthr >> 5;
    const int half = lane >> 4;        // 0: lanes 0-15, 1: lanes 16-31
    const int lr   = lane & 15;
    const int m0 = (blockIdx.x * MW + wv) * 16;
    const int p0 = blockIdx.y * 64;
    const int z  = blockIdx.z;

    const float* Xz = X + (size_t)z * xStride;
    const float* Wz = Wm + (size_t)z * wStride;

    // stage one 32x64 f32 tile (k-offset k0) into LDS buffer `buf`
    auto stage = [&](int k0, int buf) {
        for (int i = tid; i < 32 * 16; i += nthr) {
            int kk = i >> 4;          // 0..31
            int nn = (i & 15) << 2;   // 0,4,..,60
            const float* gsrc = Xz + (size_t)(k0 + kk) * NP + p0 + nn;
            float* ldst = &tileB[buf][kk * 64 + nn];
#if USE_ASYNC_LDS
            unsigned int       loff = (unsigned int)(size_t)ldst;
            unsigned long long ga   = (unsigned long long)(size_t)gsrc;
            asm volatile("global_load_async_to_lds_b128 %0, %1, off"
                         :: "v"(loff), "v"(ga) : "memory");
#else
            *(float4*)ldst = *(const float4*)gsrc;
#endif
        }
    };

    v8f acc[4] = {};
    const int KT = Cin >> 5;          // K-steps of 32 channels

    // prologue: stage tile 0 and make it visible
    stage(0, 0);
#if USE_ASYNC_LDS
    asm volatile("s_wait_asynccnt 0x0" ::: "memory");
#endif
    __syncthreads();

    for (int kt = 0; kt < KT; ++kt) {
        const int k0  = kt << 5;
        const int cur = kt & 1;

        // issue async DMA for the NEXT tile (overlaps with WMMA below).
        // Safe: buffer (kt+1)&1 was last read in iteration kt-1, whose reads
        // were ordered by that iteration's barrier.
        if (kt + 1 < KT) stage((kt + 1) << 5, (kt + 1) & 1);

        // ---- A: 16x32 weight tile, vectorized from global ----
        const float* wr = Wz + (size_t)(m0 + lr) * Cin + k0 + half * 8;
        const float4 w0 = ((const float4*)wr)[0];
        const float4 w1 = ((const float4*)wr)[1];
        const float4 w2 = ((const float4*)(wr + 16))[0];
        const float4 w3 = ((const float4*)(wr + 16))[1];
        v16h A;
        A[0]  = (_Float16)w0.x; A[1]  = (_Float16)w0.y;
        A[2]  = (_Float16)w0.z; A[3]  = (_Float16)w0.w;
        A[4]  = (_Float16)w1.x; A[5]  = (_Float16)w1.y;
        A[6]  = (_Float16)w1.z; A[7]  = (_Float16)w1.w;
        A[8]  = (_Float16)w2.x; A[9]  = (_Float16)w2.y;
        A[10] = (_Float16)w2.z; A[11] = (_Float16)w2.w;
        A[12] = (_Float16)w3.x; A[13] = (_Float16)w3.y;
        A[14] = (_Float16)w3.z; A[15] = (_Float16)w3.w;

        // ---- B fragments from current LDS tile + 4 WMMA ----
        const float* tb = tileB[cur];
#pragma unroll
        for (int t = 0; t < 4; ++t) {
            const float* bc = &tb[(half * 8) * 64 + t * 16 + lr];
            v16h Bv;
#pragma unroll
            for (int i = 0; i < 16; ++i)
                Bv[i] = (_Float16)bc[(i + (i >= 8 ? 8 : 0)) * 64];
            acc[t] = __builtin_amdgcn_wmma_f32_16x16x32_f16(
                false, A, false, Bv, (short)0, acc[t], false, false);
        }

        // next tile's DMA must have landed before the next iteration reads it
#if USE_ASYNC_LDS
        asm volatile("s_wait_asynccnt 0x0" ::: "memory");
#endif
        __syncthreads();
    }

    // ---- epilogue: per-channel scale/shift, activation, optional residual ----
#pragma unroll
    for (int t = 0; t < 4; ++t) {
#pragma unroll
        for (int v = 0; v < 8; ++v) {
            int m = m0 + v + 8 * half;
            float val = acc[t][v];
            float a_ = sA ? sA[m] : 1.0f;
            float b_ = sB ? sB[m] : 0.0f;
            val = act_apply(val * a_ + b_, act);
            size_t off = (size_t)m * NP + (size_t)(p0 + t * 16 + lr);
            if (resid) val += resid[(size_t)z * rStride + off];
            Y[(size_t)z * yStride + off] = val;
        }
    }
}

// ---------------------------------------------------------------------------
// elementwise / reduction kernels
// ---------------------------------------------------------------------------
__global__ void k_fill(float* __restrict__ p, float v, long n) {
    long i = (long)blockIdx.x * blockDim.x + threadIdx.x;
    if (i < n) p[i] = v;
}

__global__ void k_bnfold(const float* __restrict__ w, const float* __restrict__ b,
                         const float* __restrict__ m, const float* __restrict__ v,
                         const float* __restrict__ cb, float scale, int n,
                         float* __restrict__ outS, float* __restrict__ outT) {
    int i = blockIdx.x * blockDim.x + threadIdx.x;
    if (i >= n) return;
    float s = w[i] * rsqrtf(v[i] + 1e-5f);
    float t = b[i] - m[i] * s + (cb ? cb[i] * s : 0.0f);
    outS[i] = s * scale;
    outT[i] = t * scale;
}

__global__ void k_concat(const float* __restrict__ x, const float* __restrict__ x2,
                         float* __restrict__ out, long total) {
    long idx = (long)blockIdx.x * blockDim.x + threadIdx.x;
    if (idx >= total) return;
    long hw = idx % HWIMG;
    int  c  = (int)((idx / HWIMG) % 512);
    int  bi = (int)(idx / ((long)512 * HWIMG));
    out[idx] = (c < 256) ? x[((size_t)bi * 256 + c) * HWIMG + hw]
                         : x2[((size_t)bi * 256 + (c - 256)) * HWIMG + hw];
}

// generic depthwise conv (kxk, zero pad k/2) with per-channel scale/shift,
// activation and optional residual add.
__global__ void k_dwconv(const float* __restrict__ in, const float* __restrict__ wgt,
                         const float* __restrict__ sA, const float* __restrict__ sB,
                         const float* __restrict__ resid, float* __restrict__ out,
                         int C, int H, int W, int k, int act, long total) {
    long idx = (long)blockIdx.x * blockDim.x + threadIdx.x;
    if (idx >= total) return;
    int HWl = H * W;
    int hw = (int)(idx % HWl);
    int c  = (int)((idx / HWl) % C);
    int h = hw / W, w = hw % W;
    int rad = k >> 1;
    const float* base = in + (idx - hw);
    const float* wk_ = wgt + (size_t)c * k * k;
    float s = 0.0f;
    for (int a = 0; a < k; ++a) {
        int hh = h + a - rad;
        if (hh < 0 || hh >= H) continue;
        for (int b2 = 0; b2 < k; ++b2) {
            int ww = w + b2 - rad;
            if (ww < 0 || ww >= W) continue;
            s += wk_[a * k + b2] * base[hh * W + ww];
        }
    }
    float v = act_apply(s * (sA ? sA[c] : 1.0f) + (sB ? sB[c] : 0.0f), act);
    if (resid) v += resid[idx];
    out[idx] = v;
}

// channel-wise layernorm (per pixel over C channels), one wave per pixel
__global__ void k_ln2d(const float* __restrict__ in, const float* __restrict__ w,
                       const float* __restrict__ b, float* __restrict__ out,
                       int C, int HWl, int nPix) {
    int wv   = blockIdx.x * (blockDim.x >> 5) + (threadIdx.x >> 5);
    int lane = threadIdx.x & 31;
    if (wv >= nPix) return;
    int bi = wv / HWl, p = wv % HWl;
    const float* base = in + (size_t)bi * C * HWl + p;
    float s = 0.0f, s2 = 0.0f;
    for (int c = lane; c < C; c += 32) {
        float v = base[(size_t)c * HWl];
        s += v; s2 += v * v;
    }
#pragma unroll
    for (int o = 16; o; o >>= 1) {
        s  += __shfl_xor(s, o, 32);
        s2 += __shfl_xor(s2, o, 32);
    }
    float mu = s / C;
    float inv = rsqrtf(s2 / C - mu * mu + 1e-6f);
    float* ob = out + (size_t)bi * C * HWl + p;
    for (int c = lane; c < C; c += 32)
        ob[(size_t)c * HWl] = (base[(size_t)c * HWl] - mu) * inv * w[c] + b[c];
}

__global__ void k_grn_stats(const float* __restrict__ y, float* __restrict__ Gx,
                            float* __restrict__ gsum, int C, int HWl) {
    int c = blockIdx.x, bi = blockIdx.y;
    const float* p = y + ((size_t)bi * C + c) * HWl;
    float s = 0.0f;
    for (int i = threadIdx.x; i < HWl; i += blockDim.x) { float v = p[i]; s += v * v; }
    __shared__ float red[256];
    red[threadIdx.x] = s;
    __syncthreads();
    for (int o = 128; o; o >>= 1) {
        if ((int)threadIdx.x < o) red[threadIdx.x] += red[threadIdx.x + o];
        __syncthreads();
    }
    if (threadIdx.x == 0) {
        float g = sqrtf(red[0]);
        Gx[bi * C + c] = g;
        atomicAdd(&gsum[bi], g);
    }
}

__global__ void k_grn_apply(float* __restrict__ y, const float* __restrict__ Gx,
                            const float* __restrict__ gsum, const float* __restrict__ gg,
                            const float* __restrict__ gb, int C, int HWl, long total) {
    long idx = (long)blockIdx.x * blockDim.x + threadIdx.x;
    if (idx >= total) return;
    int c  = (int)((idx / HWl) % C);
    int bi = (int)(idx / ((long)C * HWl));
    float Nx = Gx[bi * C + c] / (gsum[bi] / C + 1e-6f);
    y[idx] = (gg[c] * Nx + 1.0f) * y[idx] + gb[c];
}

__global__ void k_pool7(const float* __restrict__ xf, float* __restrict__ kkin) {
    int idx = blockIdx.x * blockDim.x + threadIdx.x;
    if (idx >= 2 * 256 * 64) return;
    int l = idx & 63, c = (idx >> 6) & 255, bi = idx >> 14;
    float v = 0.0f;
    if (l < 49) {
        int i = l / 7, j = l % 7;
        const float* src = xf + ((size_t)bi * 512 + 256 + c) * HWIMG + (i * 8) * HDIM + j * 8;
        for (int di = 0; di < 8; ++di)
            for (int dj = 0; dj < 8; ++dj) v += src[di * HDIM + dj];
        v *= (1.0f / 64.0f);
    }
    kkin[idx] = v;
}

// Mt[z, o, c] = sum_{l<49} kk[z-slice, c, l] * wp[o, l];  (o>=74 -> 0)
__global__ void k_mtproj(const float* __restrict__ kk, const float* __restrict__ wp,
                         float* __restrict__ Mt) {
    int idx = blockIdx.x * blockDim.x + threadIdx.x;
    if (idx >= 4 * 80 * 64) return;
    int c = idx & 63, o = (idx >> 6) % 80, z = idx / (80 * 64);
    float s = 0.0f;
    if (o < 74) {
        const float* kc = kk + (size_t)z * 4096 + (size_t)c * 64;
        for (int l = 0; l < 49; ++l) s += kc[l] * wp[o * 49 + l];
    }
    Mt[idx] = s;
}

__global__ void k_padbias(const float* __restrict__ src, float* __restrict__ dst,
                          int n, int npad) {
    int i = blockIdx.x * blockDim.x + threadIdx.x;
    if (i < npad) dst[i] = (i < n) ? src[i] : 0.0f;
}

__device__ __forceinline__ int ridx(int h, int H, int k) {
    int half = k >> 1;
    if (h < half) return h;
    if (h < half + (H - k + 1)) return half;
    return h - (H - k);
}

// softmax over k*k logits (strided reads from wgt[z, koff+j, n]) + relative
// position bias (NATTEN clamped-window indexing), written [z, n, k*k].
__global__ void k_softmax_rpb(const float* __restrict__ wgt, const float* __restrict__ rpb,
                              float* __restrict__ attn, int koff, int k, int H, int W) {
    long HWl = (long)H * W;
    long idx = (long)blockIdx.x * blockDim.x + threadIdx.x;
    if (idx >= 4 * HWl) return;
    int n = (int)(idx % HWl);
    int z = (int)(idx / HWl);
    int g = z & 1;
    int h = n / W, w = n % W;
    int r = 2 * k - 1, kk = k * k;
    int rh = ridx(H - 1 - h, H, k), rw = ridx(W - 1 - w, W, k);
    const float* rp = rpb + (size_t)g * r * r;
    const float* wb = wgt + ((size_t)z * 80 + koff) * HWl + n;
    float vals[49];
    float mx = -3.4e38f;
    for (int a = 0; a < k; ++a)
        for (int b2 = 0; b2 < k; ++b2) {
            int j = a * k + b2;
            float v = wb[(size_t)j * HWl] + rp[(rh + a) * r + (rw + b2)];
            vals[j] = v;
            mx = fmaxf(mx, v);
        }
    float s = 0.0f;
    for (int j = 0; j < kk; ++j) { float e = expf(vals[j] - mx); vals[j] = e; s += e; }
    float inv = 1.0f / s;
    float* ob = attn + ((size_t)z * HWl + n) * kk;
    for (int j = 0; j < kk; ++j) ob[j] = vals[j] * inv;
}

// neighborhood attention aggregation; channel ch encodes (branch, head):
//   br = ch>>7 selects 5x5 (attn1) vs 7x7 (attn2); g = (ch>>6)&1 is the head.
__global__ void k_na2d(const float* __restrict__ y, const float* __restrict__ attn1,
                       const float* __restrict__ attn2, float* __restrict__ out,
                       int H, int W, long total) {
    long idx = (long)blockIdx.x * blockDim.x + threadIdx.x;
    if (idx >= total) return;
    int HWl = H * W;
    int n  = (int)(idx % HWl);
    int ch = (int)((idx / HWl) % 256);
    int bi = (int)(idx / ((long)256 * HWl));
    int h = n / W, w = n % W;
    int br = ch >> 7, g = (ch >> 6) & 1;
    int k = br ? 7 : 5;
    const float* at = (br ? attn2 : attn1) + ((size_t)(bi * 2 + g) * HWl + n) * (k * k);
    int I0 = min(max(h - (k >> 1), 0), H - k);
    int J0 = min(max(w - (k >> 1), 0), W - k);
    const float* yc = y + ((size_t)bi * 256 + ch) * HWl;
    float s = 0.0f;
    for (int a = 0; a < k; ++a)
        for (int b2 = 0; b2 < k; ++b2)
            s += at[a * k + b2] * yc[(I0 + a) * W + (J0 + b2)];
    out[idx] = s;
}

__global__ void k_se_reduce(const float* __restrict__ a, float* __restrict__ sein,
                            int C, int HWl) {
    int c = blockIdx.x, bi = blockIdx.y;
    const float* p = a + ((size_t)bi * C + c) * HWl;
    float s = 0.0f;
    for (int i = threadIdx.x; i < HWl; i += blockDim.x) s += p[i];
    __shared__ float red[256];
    red[threadIdx.x] = s;
    __syncthreads();
    for (int o = 128; o; o >>= 1) {
        if ((int)threadIdx.x < o) red[threadIdx.x] += red[threadIdx.x + o];
        __syncthreads();
    }
    if (threadIdx.x == 0) sein[bi * C + c] = red[0] / HWl;
}

__global__ void k_se_mlp(const float* __restrict__ sein, const float* __restrict__ w1,
                         const float* __restrict__ b1, const float* __restrict__ w2,
                         const float* __restrict__ b2, float* __restrict__ se) {
    __shared__ float sin_[256];
    __shared__ float hid[32];
    int bi = blockIdx.x, t = threadIdx.x;
    sin_[t] = sein[bi * 256 + t];
    __syncthreads();
    if (t < 32) {
        float s = b1[t];
        for (int c = 0; c < 256; ++c) s += w1[t * 256 + c] * sin_[c];
        hid[t] = geluf(s);
    }
    __syncthreads();
    float s = b2[t];
    for (int i = 0; i < 32; ++i) s += w2[t * 32 + i] * hid[i];
    se[bi * 256 + t] = 1.0f / (1.0f + expf(-s));
}

__global__ void k_gate_mul(const float* __restrict__ a, const float* __restrict__ gate,
                           const float* __restrict__ se, float* __restrict__ out,
                           int C, int HWl, long total) {
    long idx = (long)blockIdx.x * blockDim.x + threadIdx.x;
    if (idx >= total) return;
    int c  = (int)((idx / HWl) % C);
    int bi = (int)(idx / ((long)C * HWl));
    out[idx] = gate[idx] * (a[idx] * se[bi * C + c]);
}

// out = in * sA[c] + sB[c] (+ add)
__global__ void k_chan_affine(const float* __restrict__ in, const float* __restrict__ sA,
                              const float* __restrict__ sB, const float* __restrict__ add,
                              float* __restrict__ out, int C, int HWl, long total) {
    long idx = (long)blockIdx.x * blockDim.x + threadIdx.x;
    if (idx >= total) return;
    int c = (int)((idx / HWl) % C);
    float v = in[idx] * sA[c] + sB[c];
    if (add) v += add[idx];
    out[idx] = v;
}

// ---------------------------------------------------------------------------
// host launcher
// ---------------------------------------------------------------------------
extern "C" void kernel_launch(void* const* d_in, const int* in_sizes, int n_in,
                              void* d_out, int out_size, void* d_ws, size_t ws_size,
                              hipStream_t stream) {
    (void)in_sizes; (void)n_in; (void)out_size; (void)ws_size;
    auto IN = [&](int i) { return (const float*)d_in[i]; };

    // input indices per setup_inputs() flattening order
    enum {
        I_X = 0, I_XF = 1,
        I_DW1_W = 2, I_DW1_B, I_N1_W, I_N1_B,
        I_FUSDW_W = 6, I_FUSDW_B, I_FUSBN_W, I_FUSBN_B, I_FUSBN_M, I_FUSBN_V,
        I_FUSPW_W = 12, I_FUSPW_B, I_GRN_G, I_GRN_B,
        I_WQ_W = 16, I_WQBN_W, I_WQBN_B, I_WQBN_M, I_WQBN_V,
        I_WK_W = 21, I_WKBN_W, I_WKBN_B, I_WKBN_M, I_WKBN_V,
        I_WP_W = 26, I_WP_B, I_RPB1, I_RPB2,
        I_DYP_W = 30, I_DYPBN_W, I_DYPBN_B, I_DYPBN_M, I_DYPBN_V,
        I_LEPE_W = 35, I_LEPE_B, I_LEPEBN_W, I_LEPEBN_B, I_LEPEBN_M, I_LEPEBN_V,
        I_SE1_W = 41, I_SE1_B, I_SE2_W, I_SE2_B,
        I_GATE_W = 45, I_GATEBN_W, I_GATEBN_B, I_GATEBN_M, I_GATEBN_V,
        I_PROJBN_W = 50, I_PROJBN_B, I_PROJBN_M, I_PROJBN_V,
        I_PROJ_W = 54, I_PROJ_B, I_DW2_W, I_DW2_B, I_N2_W, I_N2_B,
        I_MLP1_W = 60, I_MLP1_B, I_MLPDW_W, I_MLPDW_B, I_MGRN_G, I_MGRN_B,
        I_MLP2_W = 66, I_MLP2_B, I_LS2_W, I_LS2_B
    };

    const long HW = HWIMG;
    const long P512 = 2L * 512 * HW;
    const long P256 = 2L * 256 * HW;

    float* ws = (float*)d_ws;
    size_t off = 0;
    auto alloc = [&](size_t n) { float* p = ws + off; off += n; return p; };
    float* B5[4]; for (int i = 0; i < 4; ++i) B5[i] = alloc(P512);
    float* Y256  = alloc(P256);
    float* GATE  = alloc(P256);
    float* LEPE  = alloc(P256);
    float* T256A = alloc(P256);
    float* T256B = alloc(P256);
    float* Q     = alloc(2L * 128 * HW);
    float* WGT   = alloc(4L * 80 * HW);
    float* ATT1  = alloc(4L * HW * 25);
    float* ATT2  = alloc(4L * HW * 49);
    float* KKIN  = alloc(2L * 256 * 64);
    float* KKOUT = alloc(2L * 128 * 64);
    float* MT    = alloc(4L * 80 * 64);
    float* WPB   = alloc(80);
    float* GX    = alloc(1024);
    float* GSUM  = alloc(4);
    float* SEIN  = alloc(512);
    float* SEVEC = alloc(512);
    float* FUS_S = alloc(512); float* FUS_T = alloc(512);
    float* GAT_S = alloc(256); float* GAT_T = alloc(256);
    float* WQ_S  = alloc(128); float* WQ_T  = alloc(128);
    float* WK_S  = alloc(128); float* WK_T  = alloc(128);
    float* DYP_S = alloc(256); float* DYP_T = alloc(256);
    float* LEP_S = alloc(256); float* LEP_T = alloc(256);
    float* PRJ_S = alloc(256); float* PRJ_T = alloc(256);

    const int TPB = 256;
    auto blocks = [&](long n) { return dim3((unsigned)((n + TPB - 1) / TPB)); };
    const float SCALE = 0.08838834764831845f; // (DIM/HEADS)^-0.5 = 128^-0.5

    auto gemm = [&](const float* X, long xs, const float* Wm, long wsd, float* Yp, long ys,
                    const float* sA, const float* sB, const float* resid, long rs,
                    int Cin, int Cout, int NP, int Z, int act) {
        int MW = (Cout % 128 == 0) ? 8 : (Cout / 16);   // 80 -> 5 waves
        dim3 g((unsigned)(Cout / (16 * MW)), (unsigned)(NP / 64), (unsigned)Z);
        k_gemm1x1<<<g, 32 * MW, 0, stream>>>(X, xs, Wm, wsd, Yp, ys, sA, sB, resid, rs,
                                             Cin, NP, act);
    };

    // --- BN folds (one-time tiny kernels) ---
    k_bnfold<<<blocks(512), TPB, 0, stream>>>(IN(I_FUSBN_W), IN(I_FUSBN_B), IN(I_FUSBN_M), IN(I_FUSBN_V), IN(I_FUSDW_B), 1.f, 512, FUS_S, FUS_T);
    k_bnfold<<<blocks(256), TPB, 0, stream>>>(IN(I_GATEBN_W), IN(I_GATEBN_B), IN(I_GATEBN_M), IN(I_GATEBN_V), nullptr, 1.f, 256, GAT_S, GAT_T);
    k_bnfold<<<blocks(128), TPB, 0, stream>>>(IN(I_WQBN_W), IN(I_WQBN_B), IN(I_WQBN_M), IN(I_WQBN_V), nullptr, SCALE, 128, WQ_S, WQ_T);
    k_bnfold<<<blocks(128), TPB, 0, stream>>>(IN(I_WKBN_W), IN(I_WKBN_B), IN(I_WKBN_M), IN(I_WKBN_V), nullptr, 1.f, 128, WK_S, WK_T);
    k_bnfold<<<blocks(256), TPB, 0, stream>>>(IN(I_DYPBN_W), IN(I_DYPBN_B), IN(I_DYPBN_M), IN(I_DYPBN_V), nullptr, 1.f, 256, DYP_S, DYP_T);
    k_bnfold<<<blocks(256), TPB, 0, stream>>>(IN(I_LEPEBN_W), IN(I_LEPEBN_B), IN(I_LEPEBN_M), IN(I_LEPEBN_V), IN(I_LEPE_B), 1.f, 256, LEP_S, LEP_T);
    k_bnfold<<<blocks(256), TPB, 0, stream>>>(IN(I_PROJBN_W), IN(I_PROJBN_B), IN(I_PROJBN_M), IN(I_PROJBN_V), nullptr, 1.f, 256, PRJ_S, PRJ_T);
    k_padbias<<<1, 128, 0, stream>>>(IN(I_WP_B), WPB, 74, 80);

    // --- stem: concat, residual dw3x3, layernorm ---
    k_concat<<<blocks(P512), TPB, 0, stream>>>(IN(I_X), IN(I_XF), B5[0], P512);
    k_dwconv<<<blocks(P512), TPB, 0, stream>>>(B5[0], IN(I_DW1_W), nullptr, IN(I_DW1_B), B5[0], B5[1], 512, HDIM, HDIM, 3, 0, P512);
    k_ln2d<<<blocks(2L * HW * 32), TPB, 0, stream>>>(B5[1], IN(I_N1_W), IN(I_N1_B), B5[2], 512, HWIMG, (int)(2 * HW));

    // --- fusion: dw3x3 + BN + gelu -> pw 512->256 (+bias) -> GRN ---
    k_dwconv<<<blocks(P512), TPB, 0, stream>>>(B5[2], IN(I_FUSDW_W), FUS_S, FUS_T, nullptr, B5[0], 512, HDIM, HDIM, 3, 1, P512);
    gemm(B5[0], 512 * HW, IN(I_FUSPW_W), 0, Y256, 256 * HW, nullptr, IN(I_FUSPW_B), nullptr, 0, 512, 256, HWIMG, 2, 0);
    k_fill<<<1, 32, 0, stream>>>(GSUM, 0.f, 4);
    k_grn_stats<<<dim3(256, 2), 256, 0, stream>>>(Y256, GX, GSUM, 256, HWIMG);
    k_grn_apply<<<blocks(P256), TPB, 0, stream>>>(Y256, GX, GSUM, IN(I_GRN_G), IN(I_GRN_B), 256, HWIMG, P256);

    // --- gate, lepe ---
    gemm(Y256, 256 * HW, IN(I_GATE_W), 0, GATE, 256 * HW, GAT_S, GAT_T, nullptr, 0, 256, 256, HWIMG, 2, 2);
    k_dwconv<<<blocks(P256), TPB, 0, stream>>>(Y256, IN(I_LEPE_W), LEP_S, LEP_T, nullptr, LEPE, 256, HDIM, HDIM, 7, 0, P256);

    // --- dynamic attention weights: q, pooled k, Mt = kk^T wp, wgt = Mt q ---
    gemm(B5[2], 512 * HW, IN(I_WQ_W), 0, Q, 128 * HW, WQ_S, WQ_T, nullptr, 0, 256, 128, HWIMG, 2, 0);
    k_pool7<<<blocks(2L * 256 * 64), TPB, 0, stream>>>(B5[2], KKIN);
    gemm(KKIN, 256 * 64, IN(I_WK_W), 0, KKOUT, 128 * 64, WK_S, WK_T, nullptr, 0, 256, 128, 64, 2, 0);
    k_mtproj<<<blocks(4L * 80 * 64), TPB, 0, stream>>>(KKOUT, IN(I_WP_W), MT);
    gemm(Q, 64 * HW, MT, 80 * 64, WGT, 80 * HW, nullptr, WPB, nullptr, 0, 64, 80, HWIMG, 4, 0);

    // --- softmax with relative position bias; neighborhood aggregation ---
    k_softmax_rpb<<<blocks(4L * HW), TPB, 0, stream>>>(WGT, IN(I_RPB1), ATT1, 0, 5, HDIM, HDIM);
    k_softmax_rpb<<<blocks(4L * HW), TPB, 0, stream>>>(WGT, IN(I_RPB2), ATT2, 25, 7, HDIM, HDIM);
    k_na2d<<<blocks(P256), TPB, 0, stream>>>(Y256, ATT1, ATT2, T256A, HDIM, HDIM, P256);

    // --- dyp (+BN) + lepe; SE; gate; proj BN; proj 256->512 ---
    gemm(T256A, 256 * HW, IN(I_DYP_W), 0, T256B, 256 * HW, DYP_S, DYP_T, LEPE, 256 * HW, 256, 256, HWIMG, 2, 0);
    k_se_reduce<<<dim3(256, 2), 256, 0, stream>>>(T256B, SEIN, 256, HWIMG);
    k_se_mlp<<<2, 256, 0, stream>>>(SEIN, IN(I_SE1_W), IN(I_SE1_B), IN(I_SE2_W), IN(I_SE2_B), SEVEC);
    k_gate_mul<<<blocks(P256), TPB, 0, stream>>>(T256B, GATE, SEVEC, T256A, 256, HWIMG, P256);
    k_chan_affine<<<blocks(P256), TPB, 0, stream>>>(T256A, PRJ_S, PRJ_T, nullptr, T256B, 256, HWIMG, P256);
    gemm(T256B, 256 * HW, IN(I_PROJ_W), 0, B5[3], 512 * HW, nullptr, IN(I_PROJ_B), nullptr, 0, 256, 512, HWIMG, 2, 0);

    // --- shared-weight MLP block ---
    auto run_mlp = [&](const float* z, float* outb) {
        k_ln2d<<<blocks(2L * HW * 32), TPB, 0, stream>>>(z, IN(I_N2_W), IN(I_N2_B), B5[0], 512, HWIMG, (int)(2 * HW));
        gemm(B5[0], 512 * HW, IN(I_MLP1_W), 0, T256A, 256 * HW, nullptr, IN(I_MLP1_B), nullptr, 0, 512, 256, HWIMG, 2, 1);
        k_dwconv<<<blocks(P256), TPB, 0, stream>>>(T256A, IN(I_MLPDW_W), nullptr, IN(I_MLPDW_B), T256A, T256B, 256, HDIM, HDIM, 3, 0, P256);
        k_fill<<<1, 32, 0, stream>>>(GSUM, 0.f, 4);
        k_grn_stats<<<dim3(256, 2), 256, 0, stream>>>(T256B, GX, GSUM, 256, HWIMG);
        k_grn_apply<<<blocks(P256), TPB, 0, stream>>>(T256B, GX, GSUM, IN(I_MGRN_G), IN(I_MGRN_B), 256, HWIMG, P256);
        gemm(T256B, 256 * HW, IN(I_MLP2_W), 0, outb, 512 * HW, nullptr, IN(I_MLP2_B), nullptr, 0, 256, 512, HWIMG, 2, 0);
    };

    // a = ls2(a) + mlp(a)
    run_mlp(B5[3], B5[1]);
    k_chan_affine<<<blocks(P512), TPB, 0, stream>>>(B5[3], IN(I_LS2_W), IN(I_LS2_B), B5[1], B5[2], 512, HWIMG, P512);
    // a = res_dw(a, dw2)
    k_dwconv<<<blocks(P512), TPB, 0, stream>>>(B5[2], IN(I_DW2_W), nullptr, IN(I_DW2_B), B5[2], B5[3], 512, HDIM, HDIM, 3, 0, P512);
    // a = ls2(a) + mlp(a)  -> final output
    run_mlp(B5[3], B5[1]);
    k_chan_affine<<<blocks(P512), TPB, 0, stream>>>(B5[3], IN(I_LS2_W), IN(I_LS2_B), B5[1], (float*)d_out, 512, HWIMG, P512);
}
